// TransformerAssociator_26173530702458
// MI455X (gfx1250) — matrix operand order repaired and checked by reading, not compiled, not run on hardware
//
#include <hip/hip_runtime.h>
#include <hip/hip_bf16.h>
#include <math.h>

typedef _Float16 f16t;
typedef _Float16 v16h __attribute__((ext_vector_type(16)));
typedef _Float16 v8h  __attribute__((ext_vector_type(8)));
typedef float    v8f  __attribute__((ext_vector_type(8)));

static inline int cdiv(int a, int b) { return (a + b - 1) / b; }

// ---------------- f32 -> f16 convert ----------------
__global__ void k_f32to16(const float* __restrict__ a, f16t* __restrict__ b, int n) {
  int i = blockIdx.x * blockDim.x + threadIdx.x;
  if (i < n) b[i] = (f16t)a[i];
}

__global__ void k_fill(float* __restrict__ p, float v, int n) {
  int i = blockIdx.x * blockDim.x + threadIdx.x;
  if (i < n) p[i] = v;
}

// ---------------- direct 3x3 conv, pad 1 ----------------
__global__ void k_conv3x3(const float* __restrict__ x, const float* __restrict__ w,
                          const float* __restrict__ bias, float* __restrict__ y,
                          int Cin, int Hin, int Win, int Cout, int Ho, int Wo,
                          int stride, int leaky) {
  int idx = blockIdx.x * blockDim.x + threadIdx.x;
  int total = Cout * Ho * Wo;
  if (idx >= total) return;
  int wo = idx % Wo, ho = (idx / Wo) % Ho, co = idx / (Wo * Ho);
  float acc = bias[co];
  const float* wk = w + (size_t)co * Cin * 9;
  for (int ci = 0; ci < Cin; ++ci) {
    const float* xp = x + (size_t)ci * Hin * Win;
    const float* wc = wk + ci * 9;
    for (int kh = 0; kh < 3; ++kh) {
      int ih = ho * stride + kh - 1;
      if (ih < 0 || ih >= Hin) continue;
      for (int kw = 0; kw < 3; ++kw) {
        int iw = wo * stride + kw - 1;
        if (iw < 0 || iw >= Win) continue;
        acc = fmaf(xp[ih * Win + iw], wc[kh * 3 + kw], acc);
      }
    }
  }
  if (leaky) acc = acc > 0.f ? acc : 0.01f * acc;
  y[idx] = acc;
}

// s[q, c] = f[c, q] + pe[c, q]   (NCHW -> (seq, d) transpose + add)
__global__ void k_combine(const float* __restrict__ f, const float* __restrict__ pe,
                          float* __restrict__ s, f16t* __restrict__ s16, int seq, int d) {
  int i = blockIdx.x * blockDim.x + threadIdx.x;
  if (i >= seq * d) return;
  int c = i % d, q = i / d;
  float v = f[c * seq + q] + pe[c * seq + q];
  s[i] = v;
  s16[i] = (f16t)v;
}

// ---------------- WMMA f16 GEMM: C[M,N] = act(scale * A @ W^T + bias) ----------------
// A: M x K f16 row-major (lda), W: N x K f16 row-major (ldw), C: f32 (ldc).
// One wave computes a 16x64 tile of C (4 accumulators share one A fragment ->
// 4 WMMAs per A-fragment load, 4x better A reuse). grid = (N/64, M/16), block = 32.
// Fragment layouts used here (wave32, contiguous k order, not the ISA table order):
//   A 16x32:  lane = m + 16*(k>>4),      elem = k & 15
//   B 32x16:  lane = n + 16*(k>>4),      elem = k & 15
//   C 16x16:  lane = n + 16*(m>>3),      vgpr r = m & 7
__global__ __launch_bounds__(32)
void k_gemm_wmma(const f16t* __restrict__ A, const f16t* __restrict__ W,
                 const float* __restrict__ bias, float* __restrict__ C,
                 f16t* __restrict__ C16, int K, int lda, int ldw, int ldc,
                 float scale, int act) {
  int lane = threadIdx.x;
  int nl = lane & 15, half = lane >> 4;
  int n0 = blockIdx.x * 64, m0 = blockIdx.y * 16;
  const f16t* arow = A + (size_t)(m0 + nl) * lda;      // row m = m0 + (lane&15)
  const f16t* w0 = W + (size_t)(n0 + nl) * ldw;        // col n = n0 + t*16 + (lane&15)
  const f16t* w1 = w0 + (size_t)16 * ldw;
  const f16t* w2 = w0 + (size_t)32 * ldw;
  const f16t* w3 = w0 + (size_t)48 * ldw;
  v8f acc0 = {}, acc1 = {}, acc2 = {}, acc3 = {};
  for (int k0 = 0; k0 < K; k0 += 32) {
    if (k0 + 32 < K) {  // prefetch next k-step (lowers to global_prefetch_b8)
      __builtin_prefetch(arow + k0 + 32, 0, 1);
      __builtin_prefetch(w0 + k0 + 32, 0, 1);
      __builtin_prefetch(w1 + k0 + 32, 0, 1);
      __builtin_prefetch(w2 + k0 + 32, 0, 1);
      __builtin_prefetch(w3 + k0 + 32, 0, 1);
    }
    v16h av = *reinterpret_cast<const v16h*>(arow + k0 + 16 * half);    // k = 16*half..+15
    v8h lo = av.lo, hi = av.hi;
    v16h a;
#pragma unroll
    for (int e = 0; e < 8; ++e) { a[e] = lo[e]; a[e + 8] = hi[e]; }
    v16h b0 = *reinterpret_cast<const v16h*>(w0 + k0 + 16 * half);      // k = 16*half..+15
    v16h b1 = *reinterpret_cast<const v16h*>(w1 + k0 + 16 * half);
    v16h b2 = *reinterpret_cast<const v16h*>(w2 + k0 + 16 * half);
    v16h b3 = *reinterpret_cast<const v16h*>(w3 + k0 + 16 * half);
    acc0 = __builtin_amdgcn_wmma_f32_16x16x32_f16(false, a, false, b0, (short)0, acc0, false, false);
    acc1 = __builtin_amdgcn_wmma_f32_16x16x32_f16(false, a, false, b1, (short)0, acc1, false, false);
    acc2 = __builtin_amdgcn_wmma_f32_16x16x32_f16(false, a, false, b2, (short)0, acc2, false, false);
    acc3 = __builtin_amdgcn_wmma_f32_16x16x32_f16(false, a, false, b3, (short)0, acc3, false, false);
  }
#pragma unroll
  for (int t = 0; t < 4; ++t) {
    v8f acc = (t == 0) ? acc0 : (t == 1) ? acc1 : (t == 2) ? acc2 : acc3;
    int n = n0 + 16 * t + nl;
    float bv = bias ? bias[n] : 0.f;
#pragma unroll
    for (int r = 0; r < 8; ++r) {
      int m = m0 + half * 8 + r;
      float v = fmaf(acc[r], scale, bv);
      if (act == 1) v = v > 0.f ? v : 0.f;            // relu
      else if (act == 2) v = v > 0.f ? v : 0.01f * v; // leaky 0.01
      C[(size_t)m * ldc + n] = v;
      if (C16) C16[(size_t)m * ldc + n] = (f16t)v;
    }
  }
}

// ---------------- multi-head attention (d=256, 8 heads, hd=32) ----------------
// Q: (L,256) f32.  KV: (L,512) f32, cols [0,256)=K, [256,512)=V.
// grid = (L, 8), block = 32. One block per (query row, head).
__global__ __launch_bounds__(32)
void k_attention(const float* __restrict__ Q, const float* __restrict__ KV,
                 float* __restrict__ O, f16t* __restrict__ O16, int L) {
  int i = blockIdx.x, h = blockIdx.y, t = threadIdx.x;
  __shared__ float sc[160];
  __shared__ float red[32];
  float qr[32];
  const float* q = Q + (size_t)i * 256 + h * 32;
#pragma unroll
  for (int d = 0; d < 32; ++d) qr[d] = q[d];
  float mx = -1e30f;
  for (int j = t; j < L; j += 32) {
    const float* kp = KV + (size_t)j * 512 + h * 32;
    float dot = 0.f;
#pragma unroll
    for (int d = 0; d < 32; ++d) dot = fmaf(qr[d], kp[d], dot);
    dot *= 0.17677669529663687f;  // 1/sqrt(32)
    sc[j] = dot;
    mx = fmaxf(mx, dot);
  }
  red[t] = mx;
  __syncthreads();
  if (t == 0) { float m = red[0]; for (int a = 1; a < 32; ++a) m = fmaxf(m, red[a]); red[0] = m; }
  __syncthreads();
  mx = red[0];
  __syncthreads();
  float sum = 0.f;
  for (int j = t; j < L; j += 32) { float e = __expf(sc[j] - mx); sc[j] = e; sum += e; }
  red[t] = sum;
  __syncthreads();
  if (t == 0) { float s = 0.f; for (int a = 0; a < 32; ++a) s += red[a]; red[0] = s; }
  __syncthreads();
  float inv = 1.f / red[0];
  float o = 0.f;
  for (int j = 0; j < L; ++j) o = fmaf(sc[j], KV[(size_t)j * 512 + 256 + h * 32 + t], o);
  o *= inv;
  O[(size_t)i * 256 + h * 32 + t] = o;
  O16[(size_t)i * 256 + h * 32 + t] = (f16t)o;
}

// ---------------- residual add + LayerNorm (d == blockDim == 256) ----------------
__global__ __launch_bounds__(256)
void k_add_ln(const float* __restrict__ a, const float* __restrict__ b,
              const float* __restrict__ g, const float* __restrict__ bb,
              float* __restrict__ out, f16t* __restrict__ out16, int d) {
  int row = blockIdx.x, t = threadIdx.x;
  __shared__ float red[256];
  float x = a[(size_t)row * d + t] + b[(size_t)row * d + t];
  red[t] = x;
  __syncthreads();
  for (int s = 128; s > 0; s >>= 1) { if (t < s) red[t] += red[t + s]; __syncthreads(); }
  float mu = red[0] / d;
  __syncthreads();
  float dv = x - mu;
  red[t] = dv * dv;
  __syncthreads();
  for (int s = 128; s > 0; s >>= 1) { if (t < s) red[t] += red[t + s]; __syncthreads(); }
  float var = red[0] / d;
  float y = dv * rsqrtf(var + 1e-5f) * g[t] + bb[t];
  out[(size_t)row * d + t] = y;
  out16[(size_t)row * d + t] = (f16t)y;
}

// ---------------- second MLP layer: (rows,hid) x (hid,) + b -> (rows,) ----------------
__global__ void k_mlp_out(const float* __restrict__ h, const float* __restrict__ w1,
                          const float* __restrict__ b1, float* __restrict__ out,
                          int rows, int hid) {
  int r = blockIdx.x * blockDim.x + threadIdx.x;
  if (r >= rows) return;
  float acc = b1[0];
  for (int k = 0; k < hid; ++k) acc = fmaf(h[(size_t)r * hid + k], w1[k], acc);
  out[r] = acc;
}

// ---------------- bilinear resize (HS,WS,C) f32 -> (HD,WD,C) f16, half-pixel ----------------
__global__ void k_resize(const float* __restrict__ s, f16t* __restrict__ d,
                         int HS, int WS, int HD, int WD, int C) {
  int idx = blockIdx.x * blockDim.x + threadIdx.x;
  int total = HD * WD * C;
  if (idx >= total) return;
  int c = idx % C, x = (idx / C) % WD, y = idx / (C * WD);
  float sy = (y + 0.5f) * ((float)HS / HD) - 0.5f;
  float sx = (x + 0.5f) * ((float)WS / WD) - 0.5f;
  float fy = sy - floorf(sy), fx = sx - floorf(sx);
  int y0 = (int)floorf(sy), x0 = (int)floorf(sx);
  int y1 = min(max(y0 + 1, 0), HS - 1), x1 = min(max(x0 + 1, 0), WS - 1);
  y0 = min(max(y0, 0), HS - 1); x0 = min(max(x0, 0), WS - 1);
  float v00 = s[((size_t)y0 * WS + x0) * C + c];
  float v01 = s[((size_t)y0 * WS + x1) * C + c];
  float v10 = s[((size_t)y1 * WS + x0) * C + c];
  float v11 = s[((size_t)y1 * WS + x1) * C + c];
  float v0 = v00 + (v01 - v00) * fx;
  float v1 = v10 + (v11 - v10) * fx;
  d[idx] = (f16t)(v0 + (v1 - v0) * fy);
}

// ---------------- append bin row/col of alpha to Z ((m+1) x (n+1)) ----------------
__global__ void k_bins(float* __restrict__ Z, const float* __restrict__ alpha, int m, int n) {
  int k = blockIdx.x * blockDim.x + threadIdx.x;
  float a = *alpha;
  int ld = n + 1;
  if (k < m) Z[(size_t)k * ld + n] = a;
  if (k < n) Z[(size_t)m * ld + k] = a;
  if (k == 0) Z[(size_t)m * ld + n] = a;
}

// ---------------- Sinkhorn row/col logsumexp update ----------------
// dst[i] = target[i] - LSE_j(Z[i,j] + other[j])   (isCol: transpose access)
__global__ __launch_bounds__(256)
void k_lse(const float* __restrict__ Z, const float* __restrict__ other,
           float* __restrict__ dst, int np1, float norm, float lastExtra, int isCol) {
  int i = blockIdx.x, t = threadIdx.x;
  __shared__ float red[256];
  float mx = -1e30f;
  for (int j = t; j < np1; j += 256) {
    float z = isCol ? Z[(size_t)j * np1 + i] : Z[(size_t)i * np1 + j];
    mx = fmaxf(mx, z + other[j]);
  }
  red[t] = mx;
  __syncthreads();
  for (int s = 128; s > 0; s >>= 1) { if (t < s) red[t] = fmaxf(red[t], red[t + s]); __syncthreads(); }
  mx = red[0];
  __syncthreads();
  float sum = 0.f;
  for (int j = t; j < np1; j += 256) {
    float z = isCol ? Z[(size_t)j * np1 + i] : Z[(size_t)i * np1 + j];
    sum += __expf(z + other[j] - mx);
  }
  red[t] = sum;
  __syncthreads();
  for (int s = 128; s > 0; s >>= 1) { if (t < s) red[t] += red[t + s]; __syncthreads(); }
  if (t == 0) {
    float target = (i < np1 - 1) ? norm : (lastExtra + norm);
    dst[i] = target - (mx + __logf(red[0]));
  }
}

__global__ void k_finalize(float* __restrict__ Z, const float* __restrict__ u,
                           const float* __restrict__ v, int np1, float norm) {
  size_t idx = (size_t)blockIdx.x * blockDim.x + threadIdx.x;
  size_t total = (size_t)np1 * np1;
  if (idx >= total) return;
  int i = (int)(idx / np1), j = (int)(idx % np1);
  Z[idx] += u[i] + v[j] - norm;
}

// =======================================================================
extern "C" void kernel_launch(void* const* d_in, const int* in_sizes, int n_in,
                              void* d_out, int out_size, void* d_ws, size_t ws_size,
                              hipStream_t stream) {
  (void)in_sizes; (void)n_in; (void)out_size; (void)ws_size;
  // setup_inputs() dict order, params in _init_params insertion order:
  const float* bgrs0 = (const float*)d_in[0];
  const float* pes0  = (const float*)d_in[1];
  const float* bgrs1 = (const float*)d_in[2];
  const float* pes1  = (const float*)d_in[3];
  const float* encw[3]  = {(const float*)d_in[4],  (const float*)d_in[5],  (const float*)d_in[6]};
  const float* encb[3]  = {(const float*)d_in[7],  (const float*)d_in[8],  (const float*)d_in[9]};
  const float* kencw[3] = {(const float*)d_in[10], (const float*)d_in[11], (const float*)d_in[12]};
  const float* kencb[3] = {(const float*)d_in[13], (const float*)d_in[14], (const float*)d_in[15]};
  const float* t_wqkv = (const float*)d_in[16];  // (4,768,256)
  const float* t_bqkv = (const float*)d_in[17];  // (4,768)
  const float* t_wo   = (const float*)d_in[18];  // (4,256,256)
  const float* t_bo   = (const float*)d_in[19];
  const float* ln1g   = (const float*)d_in[20];
  const float* ln1b   = (const float*)d_in[21];
  const float* t_w1   = (const float*)d_in[22];  // (4,1024,256)
  const float* t_b1   = (const float*)d_in[23];
  const float* t_w2   = (const float*)d_in[24];  // (4,256,1024)
  const float* t_b2   = (const float*)d_in[25];
  const float* ln2g   = (const float*)d_in[26];
  const float* ln2b   = (const float*)d_in[27];
  const float* mlp_w0 = (const float*)d_in[28];  // (128,256)
  const float* mlp_b0 = (const float*)d_in[29];
  const float* mlp_w1 = (const float*)d_in[30];  // (1,128)
  const float* mlp_b1 = (const float*)d_in[31];
  const float* alpha  = (const float*)d_in[32];  // bin_score scalar

  // ---- workspace carve-out (256B aligned so v16h loads stay 32B aligned) ----
  char* wsb = (char*)d_ws;
  size_t off = 0;
  auto carve = [&](size_t bytes) -> void* {
    void* p = wsb + off;
    off += (bytes + 255) & ~(size_t)255;
    return p;
  };
  f16t* wqkv16 = (f16t*)carve((size_t)4 * 768 * 256 * 2);
  f16t* wo16   = (f16t*)carve((size_t)4 * 256 * 256 * 2);
  f16t* w116   = (f16t*)carve((size_t)4 * 1024 * 256 * 2);
  f16t* w216   = (f16t*)carve((size_t)4 * 256 * 1024 * 2);
  f16t* mw016  = (f16t*)carve((size_t)128 * 256 * 2);
  float* tmp1  = (float*)carve((size_t)64 * 24 * 24 * 4);
  float* tmp2  = (float*)carve((size_t)128 * 12 * 12 * 4);
  float* featF = (float*)carve((size_t)256 * 144 * 4);
  float* featP = (float*)carve((size_t)256 * 144 * 4);
  float* sbuf[2]; f16t* shbuf[2]; float* snbuf[2]; f16t* snhbuf[2];
  for (int k = 0; k < 2; ++k) {
    sbuf[k]   = (float*)carve(144 * 256 * 4);
    shbuf[k]  = (f16t*)carve(144 * 256 * 2);
    snbuf[k]  = (float*)carve(144 * 256 * 4);
    snhbuf[k] = (f16t*)carve(144 * 256 * 2);
  }
  float* Qb    = (float*)carve(144 * 256 * 4);
  float* KVb   = (float*)carve(144 * 512 * 4);
  float* aoutf = (float*)carve(144 * 256 * 4);
  f16t* aout16 = (f16t*)carve(144 * 256 * 2);
  float* xb    = (float*)carve(144 * 256 * 4);
  f16t* xb16   = (f16t*)carve(144 * 256 * 2);
  float* ffb   = (float*)carve(144 * 1024 * 4);
  f16t* ff16   = (f16t*)carve(144 * 1024 * 2);
  float* pjb   = (float*)carve(144 * 256 * 4);
  float* mlph  = (float*)carve(144 * 128 * 4);
  f16t* d0h    = (f16t*)carve((size_t)2304 * 256 * 2);
  f16t* d1h    = (f16t*)carve((size_t)2304 * 256 * 2);
  float* uvec  = (float*)carve(2305 * 4);
  float* vvec  = (float*)carve(2305 * 4);

  // ---- weight conversion to f16 (every call; no cached state) ----
  auto cvt = [&](const float* src, f16t* dst, int n) {
    k_f32to16<<<cdiv(n, 256), 256, 0, stream>>>(src, dst, n);
  };
  cvt(t_wqkv, wqkv16, 4 * 768 * 256);
  cvt(t_wo,   wo16,   4 * 256 * 256);
  cvt(t_w1,   w116,   4 * 1024 * 256);
  cvt(t_w2,   w216,   4 * 256 * 1024);
  cvt(mlp_w0, mw016,  128 * 256);

  // N must be a multiple of 64 (all Ns here are: 128/256/512/1024/2304... wait 2304 = 36*64 ok)
  auto gemm = [&](const f16t* A, const f16t* W, const float* bias, float* C, f16t* C16,
                  int M, int N, int K, int lda, int ldw, int ldc, float scale, int act) {
    k_gemm_wmma<<<dim3(N / 64, M / 16), 32, 0, stream>>>(A, W, bias, C, C16,
                                                         K, lda, ldw, ldc, scale, act);
  };
  auto encode = [&](const float* img, const float* const* cw, const float* const* cb,
                    int cin0, float* out) {
    k_conv3x3<<<cdiv(64 * 24 * 24, 256), 256, 0, stream>>>(img, cw[0], cb[0], tmp1,
                                                           cin0, 48, 48, 64, 24, 24, 2, 1);
    k_conv3x3<<<cdiv(128 * 12 * 12, 256), 256, 0, stream>>>(tmp1, cw[1], cb[1], tmp2,
                                                            64, 24, 24, 128, 12, 12, 2, 1);
    k_conv3x3<<<cdiv(256 * 12 * 12, 256), 256, 0, stream>>>(tmp2, cw[2], cb[2], out,
                                                            128, 12, 12, 256, 12, 12, 1, 0);
  };

  const float NORM = -logf(4608.f);   // -log(m+n), m=n=2304
  const float LOGN = logf(2304.f);
  const size_t scoreElems = (size_t)2305 * 2305;

  for (int pair = 0; pair < 2; ++pair) {
    const float* b0 = bgrs0 + (size_t)pair * 3 * 48 * 48;
    const float* p0 = pes0  + (size_t)pair * 42 * 48 * 48;
    const float* b1 = bgrs1 + (size_t)pair * 3 * 48 * 48;
    const float* p1 = pes1  + (size_t)pair * 42 * 48 * 48;

    encode(b0, encw, encb, 3, featF);
    encode(p0, kencw, kencb, 42, featP);
    k_combine<<<cdiv(144 * 256, 256), 256, 0, stream>>>(featF, featP, sbuf[0], shbuf[0], 144, 256);
    encode(b1, encw, encb, 3, featF);
    encode(p1, kencw, kencb, 42, featP);
    k_combine<<<cdiv(144 * 256, 256), 256, 0, stream>>>(featF, featP, sbuf[1], shbuf[1], 144, 256);

    // ---- alternating self/cross transformer layers (post-norm, ReLU FFN) ----
    for (int l = 0; l < 4; ++l) {
      for (int st = 0; st < 2; ++st) {
        int kv = (l % 2 == 0) ? st : (1 - st);
        const f16t* Wl = wqkv16 + (size_t)l * 768 * 256;
        // Q = q_src @ Wq^T + bq   (144x256)
        gemm(shbuf[st], Wl, t_bqkv + l * 768, Qb, (f16t*)nullptr,
             144, 256, 256, 256, 256, 256, 1.f, 0);
        // KV = kv_src @ [Wk;Wv]^T + [bk;bv]   (144x512)
        gemm(shbuf[kv], Wl + 256 * 256, t_bqkv + l * 768 + 256, KVb, (f16t*)nullptr,
             144, 512, 256, 256, 256, 512, 1.f, 0);
        k_attention<<<dim3(144, 8), 32, 0, stream>>>(Qb, KVb, aoutf, aout16, 144);
        // out-proj
        gemm(aout16, wo16 + (size_t)l * 256 * 256, t_bo + l * 256, pjb, (f16t*)nullptr,
             144, 256, 256, 256, 256, 256, 1.f, 0);
        k_add_ln<<<144, 256, 0, stream>>>(sbuf[st], pjb, ln1g + l * 256, ln1b + l * 256,
                                          xb, xb16, 256);
        // FFN
        gemm(xb16, w116 + (size_t)l * 1024 * 256, t_b1 + l * 1024, ffb, ff16,
             144, 1024, 256, 256, 256, 1024, 1.f, 1);
        gemm(ff16, w216 + (size_t)l * 256 * 1024, t_b2 + l * 256, pjb, (f16t*)nullptr,
             144, 256, 1024, 1024, 1024, 256, 1.f, 0);
        k_add_ln<<<144, 256, 0, stream>>>(xb, pjb, ln2g + l * 256, ln2b + l * 256,
                                          snbuf[st], snhbuf[st], 256);
      }
      for (int k = 0; k < 2; ++k) {  // swap s <-> s_next (both streams updated together)
        float* tf = sbuf[k]; sbuf[k] = snbuf[k]; snbuf[k] = tf;
        f16t* th = shbuf[k]; shbuf[k] = snhbuf[k]; snhbuf[k] = th;
      }
    }

    // ---- MLP heads -> isf0s / isf1s ----
    for (int st = 0; st < 2; ++st) {
      gemm(shbuf[st], mw016, mlp_b0, mlph, (f16t*)nullptr,
           144, 128, 256, 256, 256, 128, 1.f, 2);
      float* isf = (float*)d_out + 2 * scoreElems + (size_t)st * 288 + (size_t)pair * 144;
      k_mlp_out<<<1, 256, 0, stream>>>(mlph, mlp_w1, mlp_b1, isf, 144, 128);
    }

    // ---- bilinear upsample 12x12 -> 48x48 (f16 for WMMA correlation) ----
    k_resize<<<cdiv(48 * 48 * 256, 256), 256, 0, stream>>>(sbuf[0], d0h, 12, 12, 48, 48, 256);
    k_resize<<<cdiv(48 * 48 * 256, 256), 256, 0, stream>>>(sbuf[1], d1h, 12, 12, 48, 48, 256);

    // ---- correlation GEMM (2304x2304x256) straight into Z in d_out, scaled 1/sqrt(256) ----
    float* Z = (float*)d_out + (size_t)pair * scoreElems;
    gemm(d0h, d1h, (const float*)nullptr, Z, (f16t*)nullptr,
         2304, 2304, 256, 256, 256, 2305, 0.0625f, 0);
    k_bins<<<cdiv(2305, 256), 256, 0, stream>>>(Z, alpha, 2304, 2304);

    // ---- Sinkhorn in log space (10 iters), Z lives in L2 (21 MB) ----
    k_fill<<<cdiv(2305, 256), 256, 0, stream>>>(uvec, 0.f, 2305);
    k_fill<<<cdiv(2305, 256), 256, 0, stream>>>(vvec, 0.f, 2305);
    for (int it = 0; it < 10; ++it) {
      k_lse<<<2305, 256, 0, stream>>>(Z, vvec, uvec, 2305, NORM, LOGN, 0);
      k_lse<<<2305, 256, 0, stream>>>(Z, uvec, vvec, 2305, NORM, LOGN, 1);
    }
    k_finalize<<<cdiv(2305 * 2305, 256), 256, 0, stream>>>(Z, uvec, vvec, 2305, NORM);
  }
}
